// DFineMultiscaleDeformableAttention_15942918603210
// MI455X (gfx1250) — matrix operand order repaired and checked by reading, not compiled, not run on hardware
//
#include <hip/hip_runtime.h>
#include <hip/hip_bf16.h>

// ---------------------------------------------------------------------------
// DFine multiscale deformable attention, MI455X (gfx1250, wave32, WMMA).
//
// Pipeline:
//   1) prep_query_f16 / prep_weights_f16 : fp32 -> f16, weights transposed to
//      N-major so WMMA B fragments are two 16B (8-half) loads per lane.
//   2) gemm_wmma_f16 : fused projection GEMM, M=9600, K=256, N=384
//      (cols 0..255 = sampling offsets, 256..383 = attention logits),
//      one 16x16 tile per wave, 8x v_wmma_f32_16x16x32_f16 per tile,
//      f32 accumulate + bias.
//   3) msda_sample : softmax over 16 points per (b,q,h), sampling-location
//      math, bilinear gather from value with coalesced 128B reads
//      (lane = channel), weighted accumulate, write (bs,Lq,256) output.
// ---------------------------------------------------------------------------

typedef __attribute__((ext_vector_type(16))) _Float16 v16h;
typedef __attribute__((ext_vector_type(8)))  _Float16 v8h;
typedef __attribute__((ext_vector_type(8)))  float    v8f;

#define BS        32
#define LQ        300
#define DM        256
#define NH        8
#define CHW       32          // channels per head = DM / NH
#define SUMP      16
#define LV        8500        // 80*80 + 40*40 + 20*20 + 10*10
#define NQ        (BS * LQ)   // 9600 query rows
#define NCOL_OFF  256         // H * sumP * 2
#define NCOL_ATT  128         // H * sumP
#define NCOL      (NCOL_OFF + NCOL_ATT)   // 384
#define MTILES    (NQ / 16)   // 600
#define NTILES    (NCOL / 16) // 24
#define NTILE_TOT (MTILES * NTILES)       // 14400, 8 waves/block -> 1800 blocks

// ---------------------------------------------------------------------------
// Prep: fp32 -> f16 conversions
// ---------------------------------------------------------------------------
__global__ __launch_bounds__(256) void prep_query_f16(const float* __restrict__ q,
                                                      _Float16* __restrict__ qh) {
    int i = blockIdx.x * 256 + threadIdx.x;
    if (i < NQ * DM) qh[i] = (_Float16)q[i];
}

// wT[n][k] = concat(w_off, w_attn)[k][n] as f16 (N-major for B-fragment loads)
__global__ __launch_bounds__(256) void prep_weights_f16(const float* __restrict__ w_off,
                                                        const float* __restrict__ w_attn,
                                                        _Float16* __restrict__ wT) {
    int i = blockIdx.x * 256 + threadIdx.x;
    if (i >= NCOL * DM) return;
    int n = i / DM;
    int k = i % DM;
    float v = (n < NCOL_OFF) ? w_off[k * NCOL_OFF + n]
                             : w_attn[k * NCOL_ATT + (n - NCOL_OFF)];
    wT[i] = (_Float16)v;
}

// ---------------------------------------------------------------------------
// Fused projection GEMM on the WMMA pipe.
// One wave computes one 16x16 output tile; K loop = 256/32 = 8 WMMAs.
// Fragment layouts per CDNA5 ISA 7.12.2 (wave32):
//   A 16x32 f16 : lane<16 -> M=lane, K = {0..7, 16..23}; lane>=16 -> K+8
//   B 32x16 f16 : lane<16 -> N=lane, K = {0..7, 16..23}; lane>=16 -> K+8
//   C/D 16x16 f32: VGPR r -> M = r (lanes 0-15) / r+8 (lanes 16-31), N = lane&15
// ---------------------------------------------------------------------------
__global__ __launch_bounds__(256) void gemm_wmma_f16(
    const _Float16* __restrict__ qh,      // [NQ][DM]
    const _Float16* __restrict__ wT,      // [NCOL][DM]
    const float* __restrict__ b_off,      // [NCOL_OFF]
    const float* __restrict__ b_attn,     // [NCOL_ATT]
    float* __restrict__ off_out,          // [NQ][NCOL_OFF]
    float* __restrict__ attn_out)         // [NQ][NCOL_ATT]
{
    const int wave   = threadIdx.x >> 5;
    const int lane   = threadIdx.x & 31;
    const int tile   = blockIdx.x * 8 + wave;     // exact fit: 1800*8 == 14400
    const int mt     = tile / NTILES;
    const int nt     = tile % NTILES;
    const int half16 = lane >> 4;                 // 0 or 1
    const int l15    = lane & 15;

    const _Float16* arow = qh + (size_t)(mt * 16 + l15) * DM;
    const _Float16* brow = wT + (size_t)(nt * 16 + l15) * DM;

    v8f acc = {};
#pragma unroll
    for (int k0 = 0; k0 < DM; k0 += 32) {
        // A fragment: two 16B chunks at K = k0 + half16*8 and +16
        v8h alo = *(const v8h*)(arow + k0 + half16 * 8);
        v8h ahi = *(const v8h*)(arow + k0 + half16 * 8 + 16);
        // B fragment: two 16B chunks at K = k0 + half16*8 and +16
        v8h blo = *(const v8h*)(brow + k0 + half16 * 8);
        v8h bhi = *(const v8h*)(brow + k0 + half16 * 8 + 16);
        v16h a, b;
#pragma unroll
        for (int i = 0; i < 8; ++i) {
            a[i] = alo[i]; a[i + 8] = ahi[i];
            b[i] = blo[i]; b[i + 8] = bhi[i];
        }
        acc = __builtin_amdgcn_wmma_f32_16x16x32_f16(
            /*neg_a=*/false, a, /*neg_b=*/false, b,
            /*c_mod=*/(short)0, acc, /*reuse_a=*/false, /*reuse_b=*/false);
    }

    const int n = nt * 16 + l15;
#pragma unroll
    for (int r = 0; r < 8; ++r) {
        const int m = mt * 16 + r + half16 * 8;
        if (nt < NCOL_OFF / 16) {
            off_out[(size_t)m * NCOL_OFF + n] = acc[r] + b_off[n];
        } else {
            attn_out[(size_t)m * NCOL_ATT + (n - NCOL_OFF)] =
                acc[r] + b_attn[n - NCOL_OFF];
        }
    }
}

// ---------------------------------------------------------------------------
// Softmax + sampling-location + bilinear gather + weighted sum.
// Block = 256 threads = one (b,q); thread = (head h, channel ch).
// Lanes ch=0..31 read value[b, idx, h*32+ch] -> coalesced 128B per corner.
// loc = ref_xy + off * (1/numP=0.25) * ref_wh * OFFSET_SCALE(0.5)
//     = ref_xy + off * 0.125 * ref_wh ;  pixel x = loc.x*W - 0.5 (same for y).
// ---------------------------------------------------------------------------
__global__ __launch_bounds__(256) void msda_sample(
    const float* __restrict__ refp,    // [NQ][4]  (x,y,w,h)
    const float* __restrict__ value,   // [BS][LV][DM]
    const float* __restrict__ off,     // [NQ][NCOL_OFF]
    const float* __restrict__ attnl,   // [NQ][NCOL_ATT]
    float* __restrict__ out)           // [NQ][DM]
{
    const int bq = blockIdx.x;
    const int h  = threadIdx.x >> 5;
    const int ch = threadIdx.x & 31;
    const int b  = bq / LQ;

    const float rx = refp[bq * 4 + 0];
    const float ry = refp[bq * 4 + 1];
    const float rw = refp[bq * 4 + 2];
    const float rh = refp[bq * 4 + 3];

    // softmax over the 16 points of this head (broadcast reads across lanes)
    const float* lg = attnl + (size_t)bq * NCOL_ATT + h * SUMP;
    float mx = lg[0];
#pragma unroll
    for (int p = 1; p < SUMP; ++p) mx = fmaxf(mx, lg[p]);
    float wgt[SUMP];
    float s = 0.f;
#pragma unroll
    for (int p = 0; p < SUMP; ++p) { wgt[p] = __expf(lg[p] - mx); s += wgt[p]; }
    const float inv = 1.f / s;

    const float* ob = off + (size_t)bq * NCOL_OFF + h * (SUMP * 2);
    const float* vb = value + (size_t)b * LV * DM + h * CHW + ch;

    const int lw[4]     = {80, 40, 20, 10};       // square levels
    const int lstart[4] = {0, 6400, 8000, 8400};

    float acc = 0.f;
#pragma unroll
    for (int p = 0; p < SUMP; ++p) {
        const int lvl  = p >> 2;                  // NUM_POINTS = [4,4,4,4]
        const int W    = lw[lvl];
        const int base = lstart[lvl];

        const float x = (rx + ob[2 * p + 0] * 0.125f * rw) * (float)W - 0.5f;
        const float y = (ry + ob[2 * p + 1] * 0.125f * rh) * (float)W - 0.5f;
        const float x0f = floorf(x), y0f = floorf(y);
        const int   x0  = (int)x0f,  y0  = (int)y0f;
        const float fx  = x - x0f,   fy  = y - y0f;

        float sv = 0.f;
#pragma unroll
        for (int dy = 0; dy < 2; ++dy) {
#pragma unroll
            for (int dx = 0; dx < 2; ++dx) {
                const int   xi = x0 + dx;
                const int   yi = y0 + dy;
                const float wc = (dx ? fx : 1.f - fx) * (dy ? fy : 1.f - fy);
                if (xi >= 0 && xi < W && yi >= 0 && yi < W) {
                    sv += wc * vb[(size_t)(base + yi * W + xi) * DM];
                }
            }
        }
        acc += wgt[p] * inv * sv;
    }
    out[(size_t)bq * DM + h * CHW + ch] = acc;
}

// ---------------------------------------------------------------------------
// Host launch
// ---------------------------------------------------------------------------
extern "C" void kernel_launch(void* const* d_in, const int* in_sizes, int n_in,
                              void* d_out, int out_size, void* d_ws, size_t ws_size,
                              hipStream_t stream) {
    const float* query  = (const float*)d_in[0];   // [32,300,256]
    const float* refp   = (const float*)d_in[1];   // [32,300,1,4]
    const float* value  = (const float*)d_in[2];   // [32,8500,256]
    const float* w_off  = (const float*)d_in[3];   // [256,256]
    const float* b_off  = (const float*)d_in[4];   // [256]
    const float* w_attn = (const float*)d_in[5];   // [256,128]
    const float* b_attn = (const float*)d_in[6];   // [128]
    float* out = (float*)d_out;                    // [32,300,256]

    // workspace layout (all 16B-aligned): ~19.9 MB total
    char* ws = (char*)d_ws;
    float*    off_out  = (float*)ws;                 ws += (size_t)NQ * NCOL_OFF * sizeof(float);
    float*    attn_out = (float*)ws;                 ws += (size_t)NQ * NCOL_ATT * sizeof(float);
    _Float16* qh       = (_Float16*)ws;              ws += (size_t)NQ * DM * sizeof(_Float16);
    _Float16* wT       = (_Float16*)ws;

    prep_query_f16<<<(NQ * DM + 255) / 256, 256, 0, stream>>>(query, qh);
    prep_weights_f16<<<(NCOL * DM + 255) / 256, 256, 0, stream>>>(w_off, w_attn, wT);
    gemm_wmma_f16<<<NTILE_TOT / 8, 256, 0, stream>>>(qh, wT, b_off, b_attn,
                                                     off_out, attn_out);
    msda_sample<<<NQ, 256, 0, stream>>>(refp, value, off_out, attn_out, out);
}
